// LayoutModel_50989851738530
// MI455X (gfx1250) — compile-verified, matrix operand order checked by reading, not run on hardware
//
#include <hip/hip_runtime.h>
#include <hip/hip_bf16.h>

#define CC     16
#define NN     10000
#define NCFG   2000
#define EE     50000
#define DDIM   256
#define RDIM   32
#define MM     (CC * NN)          // 160000 rows
#define INDIM  234
#define EPSV   1e-5f
#define NCHUNK 50                 // 10000 / 50 = 200 rows per stats chunk

typedef __attribute__((ext_vector_type(16))) __bf16 v16bf;
typedef __attribute__((ext_vector_type(8)))  __bf16 v8bf;
typedef __attribute__((ext_vector_type(8)))  float  v8f;
typedef __attribute__((ext_vector_type(4)))  float  v4f;

// ---------------------------------------------------------------------------
// Weight prep: cast+transpose big weights to bf16 [Dout=256, K=256] (lin1 K
// zero-padded 234->256). Matrix ids: 0=lin1, 1=lin2, 2..4=sage_Wl, 5..7=sage_Wr
// ---------------------------------------------------------------------------
__global__ void k_prep_weights(const float* __restrict__ lin1W,
                               const float* __restrict__ lin2W,
                               const float* __restrict__ sageWl,
                               const float* __restrict__ sageWr,
                               __bf16* __restrict__ wt) {
    int idx = blockIdx.x * 256 + threadIdx.x;   // 8 * 65536 total
    int m = idx >> 16;
    int r = idx & 0xFFFF;
    int n = r >> 8;      // output channel (row of transposed matrix)
    int k = r & 255;     // K index
    const float* src;
    int kdim = DDIM;
    if (m == 0)      { src = lin1W; kdim = INDIM; }
    else if (m == 1) { src = lin2W; }
    else if (m < 5)  { src = sageWl + (m - 2) * DDIM * DDIM; }
    else             { src = sageWr + (m - 5) * DDIM * DDIM; }
    float v = (k < kdim) ? src[k * DDIM + n] : 0.0f;
    wt[idx] = (__bf16)v;
}

__global__ void k_invmap(const int* __restrict__ ncid, int* __restrict__ inv) {
    int i = blockIdx.x * 256 + threadIdx.x;
    if (i < NCFG) inv[ncid[i]] = i;
}

// ---------------------------------------------------------------------------
// Feature build: out[M, 256] f32, cols 234..255 zero-padded
// ---------------------------------------------------------------------------
__global__ void k_build_feat(const float* __restrict__ xfeat,
                             const int*   __restrict__ xcfg,
                             const int*   __restrict__ xlay,
                             const int*   __restrict__ xop,
                             const int*   __restrict__ inv,
                             const float* __restrict__ embop,
                             const float* __restrict__ emblay,
                             float* __restrict__ out) {
    int row = blockIdx.x;            // c*NN + n
    int c = row / NN;
    int n = row - c * NN;
    int d = threadIdx.x;
    float v = 0.0f;
    if (d < 134) {
        v = xfeat[n * 134 + d];
    } else if (d < 206) {
        int t = (d - 134) >> 2, q = (d - 134) & 3;
        int p = inv[n];
        int val = (p >= 0) ? (xcfg[(c * NCFG + p) * 18 + t] + 2) : 0;
        v = emblay[val * 4 + q];
    } else if (d < 230) {
        int j = (d - 206) >> 2, q = (d - 206) & 3;
        int val = xlay[n * 6 + j] + 2;
        v = emblay[val * 4 + q];
    } else if (d < 234) {
        v = embop[xop[n] * 4 + (d - 230)];
    }
    out[row * DDIM + d] = v;
}

// ---------------------------------------------------------------------------
// WMMA GEMM: Out[M,256] = A1[M,256](f32->bf16) @ W1t' + bias
//            (+ A2 @ W2t' if A2 != null)
// Wt stored transposed: Wt[n, k] row-major, bf16.
// Each wave: 32-row x 64-col tile; two A fragments share 4 B fragments;
// B fragments double-buffered across K steps to hide load latency.
// ---------------------------------------------------------------------------
__device__ __forceinline__ v16bf load_a_frag(const float* arow, int kb, int half) {
    v4f lo0 = *(const v4f*)(arow + kb + half * 8);
    v4f lo1 = *(const v4f*)(arow + kb + half * 8 + 4);
    v4f hi0 = *(const v4f*)(arow + kb + 16 + half * 8);
    v4f hi1 = *(const v4f*)(arow + kb + 16 + half * 8 + 4);
    v16bf a;
#pragma unroll
    for (int e = 0; e < 4; e++) {
        a[e]      = (__bf16)lo0[e];
        a[e + 4]  = (__bf16)lo1[e];
        a[e + 8]  = (__bf16)hi0[e];
        a[e + 12] = (__bf16)hi1[e];
    }
    return a;
}

__device__ __forceinline__ v16bf load_b_frag(const __bf16* wrow, int kb, int half) {
    v8bf blo = *(const v8bf*)(wrow + kb + half * 8);
    v8bf bhi = *(const v8bf*)(wrow + kb + 16 + half * 8);
    v16bf b;
#pragma unroll
    for (int e = 0; e < 8; e++) {
        b[e]     = blo[e];
        b[e + 8] = bhi[e];
    }
    return b;
}

__global__ __launch_bounds__(256)
void k_gemm(const float*  __restrict__ A1,
            const __bf16* __restrict__ W1t,
            const float*  __restrict__ bias,
            const float*  __restrict__ A2,
            const __bf16* __restrict__ W2t,
            float* __restrict__ Out) {
    int wave = threadIdx.x >> 5;
    int lane = threadIdx.x & 31;
    int tile = blockIdx.x * 8 + wave;      // (M/32)*4 = 20000 tiles
    int mt = tile >> 2;
    int nt = tile & 3;
    int m0 = mt * 32;
    int n0 = nt * 64;
    int half = lane >> 4;
    int l16  = lane & 15;

    const float* a1row0 = A1 + (m0 + l16) * DDIM;
    const float* a1row1 = A1 + (m0 + 16 + l16) * DDIM;
    const __bf16* wb1[4];
#pragma unroll
    for (int t = 0; t < 4; t++) wb1[t] = W1t + (n0 + t * 16 + l16) * DDIM;

    v8f acc[8];
#pragma unroll
    for (int t = 0; t < 8; t++)
#pragma unroll
        for (int e = 0; e < 8; e++) acc[t][e] = 0.0f;

    v16bf bcur[4], bnxt[4];
#pragma unroll
    for (int t = 0; t < 4; t++) bcur[t] = load_b_frag(wb1[t], 0, half);

#pragma unroll
    for (int ks = 0; ks < 8; ks++) {
        int kb = ks * 32;
        v16bf a0 = load_a_frag(a1row0, kb, half);
        v16bf a1 = load_a_frag(a1row1, kb, half);
        if (ks < 7) {
#pragma unroll
            for (int t = 0; t < 4; t++) bnxt[t] = load_b_frag(wb1[t], kb + 32, half);
        }
#pragma unroll
        for (int t = 0; t < 4; t++) {
            acc[t]     = __builtin_amdgcn_wmma_f32_16x16x32_bf16(
                false, a0, false, bcur[t], (short)0, acc[t], false, false);
            acc[4 + t] = __builtin_amdgcn_wmma_f32_16x16x32_bf16(
                false, a1, false, bcur[t], (short)0, acc[4 + t], false, false);
        }
#pragma unroll
        for (int t = 0; t < 4; t++) bcur[t] = bnxt[t];
    }

    if (A2) {
        const float* a2row0 = A2 + (m0 + l16) * DDIM;
        const float* a2row1 = A2 + (m0 + 16 + l16) * DDIM;
        const __bf16* wb2[4];
#pragma unroll
        for (int t = 0; t < 4; t++) wb2[t] = W2t + (n0 + t * 16 + l16) * DDIM;
#pragma unroll
        for (int t = 0; t < 4; t++) bcur[t] = load_b_frag(wb2[t], 0, half);
#pragma unroll
        for (int ks = 0; ks < 8; ks++) {
            int kb = ks * 32;
            v16bf a0 = load_a_frag(a2row0, kb, half);
            v16bf a1 = load_a_frag(a2row1, kb, half);
            if (ks < 7) {
#pragma unroll
                for (int t = 0; t < 4; t++) bnxt[t] = load_b_frag(wb2[t], kb + 32, half);
            }
#pragma unroll
            for (int t = 0; t < 4; t++) {
                acc[t]     = __builtin_amdgcn_wmma_f32_16x16x32_bf16(
                    false, a0, false, bcur[t], (short)0, acc[t], false, false);
                acc[4 + t] = __builtin_amdgcn_wmma_f32_16x16x32_bf16(
                    false, a1, false, bcur[t], (short)0, acc[4 + t], false, false);
            }
#pragma unroll
            for (int t = 0; t < 4; t++) bcur[t] = bnxt[t];
        }
    }

    // C/D layout: lanes 0-15 hold M=0..7 in vgprs 0..7; lanes 16-31 hold M=8..15
#pragma unroll
    for (int t = 0; t < 4; t++) {
        int col = n0 + t * 16 + l16;
        float bv = bias[col];
#pragma unroll
        for (int r = 0; r < 8; r++) {
            Out[(m0 + half * 8 + r) * DDIM + col]      = acc[t][r] + bv;
            Out[(m0 + 16 + half * 8 + r) * DDIM + col] = acc[4 + t][r] + bv;
        }
    }
}

// ---------------------------------------------------------------------------
// Instance-norm statistics: chunked partial sums + atomics into S (sum) / Q (sumsq)
// ---------------------------------------------------------------------------
__global__ void k_stats(const float* __restrict__ X, float* __restrict__ S,
                        float* __restrict__ Q) {
    int c  = blockIdx.x / NCHUNK;
    int ch = blockIdx.x % NCHUNK;
    int d  = threadIdx.x;
    int n0 = ch * (NN / NCHUNK);
    float s = 0.0f, q = 0.0f;
    for (int i = 0; i < NN / NCHUNK; i++) {
        float v = X[(c * NN + n0 + i) * DDIM + d];
        s += v;
        q += v * v;
    }
    atomicAdd(&S[c * DDIM + d], s);
    atomicAdd(&Q[c * DDIM + d], q);
}

// ---------------------------------------------------------------------------
// Fused per-row: [instance-norm] -> [channel-attn] -> [+shortcut] -> [gelu]
// ---------------------------------------------------------------------------
__global__ __launch_bounds__(256)
void k_apply(const float* __restrict__ in, float* __restrict__ out,
             const float* __restrict__ S, const float* __restrict__ Q, int doNorm,
             const float* __restrict__ W1, const float* __restrict__ b1,
             const float* __restrict__ W2, const float* __restrict__ b2,
             const float* __restrict__ shortcut, int doGelu) {
    __shared__ float xs[DDIM];
    __shared__ float ts[RDIM];
    int row = blockIdx.x;
    int c = row / NN;
    int d = threadIdx.x;
    float x = in[row * DDIM + d];
    if (doNorm) {
        float mu  = S[c * DDIM + d] * (1.0f / NN);
        float var = Q[c * DDIM + d] * (1.0f / NN) - mu * mu;
        x = (x - mu) * rsqrtf(var + EPSV);
    }
    if (W1) {
        xs[d] = x;
        __syncthreads();
        if (d < RDIM) {
            float t = b1[d];
            for (int k = 0; k < DDIM; k++) t += xs[k] * W1[k * RDIM + d];
            ts[d] = t > 0.0f ? t : 0.0f;
        }
        __syncthreads();
        float sv = b2[d];
        for (int j = 0; j < RDIM; j++) sv += ts[j] * W2[j * DDIM + d];
        x = x * (1.0f / (1.0f + expf(-sv)));
    }
    if (shortcut) x += shortcut[row * DDIM + d];
    if (doGelu)   x = 0.5f * x * (1.0f + erff(x * 0.70710678118654752f));
    out[row * DDIM + d] = x;
}

// ---------------------------------------------------------------------------
// Edge degree count, scatter-mean aggregation
// ---------------------------------------------------------------------------
__global__ void k_count(const int* __restrict__ tgt, float* __restrict__ cnt) {
    int e = blockIdx.x * 256 + threadIdx.x;
    if (e < EE) atomicAdd(&cnt[tgt[e]], 1.0f);
}

__global__ void k_scatter(const float* __restrict__ X, const int* __restrict__ src,
                          const int* __restrict__ tgt, float* __restrict__ Agg) {
    int e = blockIdx.x;
    int d = threadIdx.x;
    int s = src[e], t = tgt[e];
    const float* xp = X + s * DDIM + d;
    float*       ap = Agg + t * DDIM + d;
#pragma unroll 4
    for (int c = 0; c < CC; c++) {
        atomicAdd(ap + c * (NN * DDIM), xp[c * (NN * DDIM)]);
    }
}

__global__ void k_scale(float* __restrict__ Agg, const float* __restrict__ cnt) {
    int row = blockIdx.x;
    int d = threadIdx.x;
    int n = row % NN;
    float cv = cnt[n];
    cv = cv > 1.0f ? cv : 1.0f;
    Agg[row * DDIM + d] *= (1.0f / cv);
}

// ---------------------------------------------------------------------------
// Mean pool over nodes + classifier
// ---------------------------------------------------------------------------
__global__ void k_pool(const float* __restrict__ X, float* __restrict__ pooled) {
    int c  = blockIdx.x / NCHUNK;
    int ch = blockIdx.x % NCHUNK;
    int d  = threadIdx.x;
    int n0 = ch * (NN / NCHUNK);
    float s = 0.0f;
    for (int i = 0; i < NN / NCHUNK; i++)
        s += X[(c * NN + n0 + i) * DDIM + d];
    atomicAdd(&pooled[c * DDIM + d], s);
}

__global__ void k_final(const float* __restrict__ pooled,
                        const float* __restrict__ clsW,
                        const float* __restrict__ clsb,
                        float* __restrict__ out) {
    __shared__ float sh[DDIM];
    int c = blockIdx.x;
    int d = threadIdx.x;
    sh[d] = pooled[c * DDIM + d] * clsW[d];
    __syncthreads();
    for (int st = 128; st > 0; st >>= 1) {
        if (d < st) sh[d] += sh[d + st];
        __syncthreads();
    }
    if (d == 0) out[c] = sh[0] * (1.0f / NN) + clsb[0];
}

// ---------------------------------------------------------------------------
extern "C" void kernel_launch(void* const* d_in, const int* in_sizes, int n_in,
                              void* d_out, int out_size, void* d_ws, size_t ws_size,
                              hipStream_t stream) {
    (void)in_sizes; (void)n_in; (void)out_size; (void)ws_size;
    // Inputs (setup_inputs order)
    const int*   x_node_cfg  = (const int*)  d_in[0];
    const float* x_feat      = (const float*)d_in[1];
    const int*   x_layout    = (const int*)  d_in[2];
    const int*   x_op        = (const int*)  d_in[3];
    const int*   edge_index  = (const int*)  d_in[4];
    const int*   node_cfg_id = (const int*)  d_in[5];
    const float* emb_op      = (const float*)d_in[6];
    const float* emb_layout  = (const float*)d_in[7];
    const float* lin1_W      = (const float*)d_in[8];
    const float* lin1_b      = (const float*)d_in[9];
    const float* ca1_W1      = (const float*)d_in[10];
    const float* ca1_b1      = (const float*)d_in[11];
    const float* ca1_W2      = (const float*)d_in[12];
    const float* ca1_b2      = (const float*)d_in[13];
    const float* lin2_W      = (const float*)d_in[14];
    const float* lin2_b      = (const float*)d_in[15];
    const float* ca2_W1      = (const float*)d_in[16];
    const float* ca2_b1      = (const float*)d_in[17];
    const float* ca2_W2      = (const float*)d_in[18];
    const float* ca2_b2      = (const float*)d_in[19];
    const float* sage_Wl     = (const float*)d_in[20];
    const float* sage_bl     = (const float*)d_in[21];
    const float* sage_Wr     = (const float*)d_in[22];
    const float* sca_W1      = (const float*)d_in[23];
    const float* sca_b1      = (const float*)d_in[24];
    const float* sca_W2      = (const float*)d_in[25];
    const float* sca_b2      = (const float*)d_in[26];
    const float* cls_W       = (const float*)d_in[27];
    const float* cls_b       = (const float*)d_in[28];
    float* out = (float*)d_out;

    const int* e_src = edge_index;
    const int* e_tgt = edge_index + EE;

    // Workspace carve-out
    char* cur = (char*)d_ws;
    auto carve = [&](size_t bytes) -> void* {
        void* p = (void*)cur;
        cur += (bytes + 255) & ~(size_t)255;
        return p;
    };
    const size_t BIG = (size_t)MM * DDIM * sizeof(float);   // 163.84 MB each
    float*  B0     = (float*)carve(BIG);
    float*  B1     = (float*)carve(BIG);
    float*  B2     = (float*)carve(BIG);
    float*  B3     = (float*)carve(BIG);
    float*  statS  = (float*)carve(CC * DDIM * sizeof(float));
    float*  statQ  = (float*)carve(CC * DDIM * sizeof(float));
    float*  pooled = (float*)carve(CC * DDIM * sizeof(float));
    float*  cnt    = (float*)carve(NN * sizeof(float));
    int*    inv    = (int*)  carve(NN * sizeof(int));
    __bf16* wt     = (__bf16*)carve((size_t)8 * DDIM * DDIM * sizeof(__bf16));

    __bf16* lin1t = wt;
    __bf16* lin2t = wt + 1 * DDIM * DDIM;
    auto Wlt = [&](int i) { return wt + (2 + i) * DDIM * DDIM; };
    auto Wrt = [&](int i) { return wt + (5 + i) * DDIM * DDIM; };

    const int GEMM_GRID = (MM / 32) * 4 / 8;   // 2500 blocks of 8 waves

    // 1. weights -> bf16 transposed
    k_prep_weights<<<(8 * DDIM * DDIM) / 256, 256, 0, stream>>>(
        lin1_W, lin2_W, sage_Wl, sage_Wr, wt);

    // 2. inverse config-id map
    hipMemsetAsync(inv, 0xFF, NN * sizeof(int), stream);
    k_invmap<<<(NCFG + 255) / 256, 256, 0, stream>>>(node_cfg_id, inv);

    // 3. feature build -> B0
    k_build_feat<<<MM, 256, 0, stream>>>(x_feat, x_node_cfg, x_layout, x_op, inv,
                                         emb_op, emb_layout, B0);

    // 4. LinearActNorm 1: B1 = B0 @ W1 + b ; IN ; attn ; gelu
    k_gemm<<<GEMM_GRID, 256, 0, stream>>>(B0, lin1t, lin1_b, nullptr, nullptr, B1);
    hipMemsetAsync(statS, 0, CC * DDIM * sizeof(float), stream);
    hipMemsetAsync(statQ, 0, CC * DDIM * sizeof(float), stream);
    k_stats<<<CC * NCHUNK, 256, 0, stream>>>(B1, statS, statQ);
    k_apply<<<MM, 256, 0, stream>>>(B1, B1, statS, statQ, 1,
                                    ca1_W1, ca1_b1, ca1_W2, ca1_b2, nullptr, 1);

    // 5. LinearActNorm 2: B0 = B1 @ W2 + b ; IN ; attn ; gelu
    k_gemm<<<GEMM_GRID, 256, 0, stream>>>(B1, lin2t, lin2_b, nullptr, nullptr, B0);
    hipMemsetAsync(statS, 0, CC * DDIM * sizeof(float), stream);
    hipMemsetAsync(statQ, 0, CC * DDIM * sizeof(float), stream);
    k_stats<<<CC * NCHUNK, 256, 0, stream>>>(B0, statS, statQ);
    k_apply<<<MM, 256, 0, stream>>>(B0, B0, statS, statQ, 1,
                                    ca2_W1, ca2_b1, ca2_W2, ca2_b2, nullptr, 1);

    // 6. edge degree counts
    hipMemsetAsync(cnt, 0, NN * sizeof(float), stream);
    k_count<<<(EE + 255) / 256, 256, 0, stream>>>(e_tgt, cnt);

    // 7. SAGE blocks
    for (int i = 0; i < 3; i++) {
        // x_n = IN(h) -> B1
        hipMemsetAsync(statS, 0, CC * DDIM * sizeof(float), stream);
        hipMemsetAsync(statQ, 0, CC * DDIM * sizeof(float), stream);
        k_stats<<<CC * NCHUNK, 256, 0, stream>>>(B0, statS, statQ);
        k_apply<<<MM, 256, 0, stream>>>(B0, B1, statS, statQ, 1,
                                        nullptr, nullptr, nullptr, nullptr, nullptr, 0);
        // agg = scatter-mean(x_n) -> B2
        hipMemsetAsync(B2, 0, BIG, stream);
        k_scatter<<<EE, 256, 0, stream>>>(B1, e_src, e_tgt, B2);
        k_scale<<<MM, 256, 0, stream>>>(B2, cnt);
        // B3 = agg @ Wl + bl + x_n @ Wr
        k_gemm<<<GEMM_GRID, 256, 0, stream>>>(B2, Wlt(i), sage_bl + i * DDIM,
                                              B1, Wrt(i), B3);
        // h = gelu(attn(B3) + shortcut) -> B0
        k_apply<<<MM, 256, 0, stream>>>(B3, B0, nullptr, nullptr, 0,
                                        sca_W1 + i * DDIM * RDIM, sca_b1 + i * RDIM,
                                        sca_W2 + i * RDIM * DDIM, sca_b2 + i * DDIM,
                                        B0, 1);
    }

    // 8. pool + classify
    hipMemsetAsync(pooled, 0, CC * DDIM * sizeof(float), stream);
    k_pool<<<CC * NCHUNK, 256, 0, stream>>>(B0, pooled);
    k_final<<<CC, 256, 0, stream>>>(pooled, cls_W, cls_b, out);
}
